// ShiftBlock_3942779978437
// MI455X (gfx1250) — matrix operand order repaired and checked by reading, not compiled, not run on hardware
//
#include <hip/hip_runtime.h>
#include <hip/hip_bf16.h>
#include <math.h>

// ---------------------------------------------------------------------------
// ShiftBlock for MI455X (gfx1250): bf16 WMMA pipeline, f32 accumulate,
// TDM (tensor_load_to_lds) double-buffered A-tile staging when available.
// ---------------------------------------------------------------------------

#define BDIM 256          // 8 waves of 32
#define BM   128          // block rows
#define BN   64           // block cols
#define BK   32           // K chunk (one wmma K step)
#define LDSA_STRIDE 40    // 32 data + 8 pad bf16 (TDM pad: 4 DW per 16 DW)
#define LDSB_STRIDE 40

typedef __attribute__((ext_vector_type(16))) __bf16 v16bf;
typedef __attribute__((ext_vector_type(8)))  __bf16 v8bf;
typedef __attribute__((ext_vector_type(8)))  float  v8f;
typedef unsigned int __attribute__((ext_vector_type(4))) uint4v;

#if defined(__has_builtin)
#if __has_builtin(__builtin_amdgcn_tensor_load_to_lds)
#define HAVE_TDM 1
#if __has_include(<hip/amd_detail/amd_gfx1250_TDM.h>)
#define TDM_6ARG 1
#endif
#endif
#endif

static __device__ __forceinline__ __bf16 f2bf(float f) {
  unsigned int u = __float_as_uint(f);
  unsigned int r = (u + 0x7fffu + ((u >> 16) & 1u)) >> 16;   // RNE
  unsigned short s = (unsigned short)r;
  return __builtin_bit_cast(__bf16, s);
}
static __device__ __forceinline__ float bf2f(__bf16 h) {
  unsigned short s = __builtin_bit_cast(unsigned short, h);
  return __uint_as_float(((unsigned int)s) << 16);
}

// ---------------------------------------------------------------------------
// TDM: DMA a bf16 tile (up to 3D) global->LDS with row padding to LDSA_STRIDE.
// Descriptor layout per CDNA5 ISA.
// pad_interval=3 (16 DWORDs = 32 bf16), pad_amount=3 (4 DWORDs = 8 bf16).
// ---------------------------------------------------------------------------
#if defined(HAVE_TDM)
typedef unsigned int __attribute__((ext_vector_type(4))) tdm_u32x4;
typedef int __attribute__((ext_vector_type(4))) tdm_i32x4;
typedef int __attribute__((ext_vector_type(8))) tdm_i32x8;

static __device__ __forceinline__ void tdm_load_bf16(
    unsigned lds_off, const __bf16* g,
    unsigned td0, unsigned td1, unsigned td2,
    unsigned long long s0, unsigned long long s1,
    unsigned t0, unsigned t1, unsigned t2) {
  unsigned long long ga = (unsigned long long)(uintptr_t)g;
  tdm_u32x4 g0;
  g0[0] = 1u;                                             // count=1, user mode
  g0[1] = lds_off;                                        // lds_addr (bytes)
  g0[2] = (unsigned)(ga & 0xffffffffu);                   // global_addr[31:0]
  g0[3] = (unsigned)((ga >> 32) & 0x1ffffffu) | (2u << 30); // addr[56:32]|type=2
  tdm_i32x8 g1;
  g1[0] = (int)((1u << 16) | (1u << 20) | (3u << 22) | (3u << 25)); // 2B,pad
  g1[1] = (int)((td0 & 0xffffu) << 16);                   // abar=0 | td0 lo16
  g1[2] = (int)(((td0 >> 16) & 0xffffu) | ((td1 & 0xffffu) << 16));
  g1[3] = (int)(((td1 >> 16) & 0xffffu) | ((t0 & 0xffffu) << 16));
  g1[4] = (int)((t1 & 0xffffu) | ((t2 & 0xffffu) << 16));
  g1[5] = (int)(unsigned)(s0 & 0xffffffffu);
  g1[6] = (int)((unsigned)((s0 >> 32) & 0xffffu) |
                ((unsigned)(s1 & 0xffffu) << 16));
  g1[7] = (int)(unsigned)((s1 >> 16) & 0xffffffffu);
  tdm_i32x4 g2;
  g2[0] = (int)td2; g2[1] = 0; g2[2] = 0; g2[3] = 0;
  tdm_i32x4 g3;
  g3[0] = 0; g3[1] = 0; g3[2] = 0; g3[3] = 0;
#if defined(TDM_6ARG)
  tdm_i32x8 z8 = {0, 0, 0, 0, 0, 0, 0, 0};
  __builtin_amdgcn_tensor_load_to_lds(g0, g1, g2, g3, z8, 0);
#else
  __builtin_amdgcn_tensor_load_to_lds(g0, g1, g2, g3, 0);
#endif
}
static __device__ __forceinline__ unsigned lds_off_of(const void* p) {
  return (unsigned)(unsigned long long)(uintptr_t)p;      // low 32 = LDS offset
}
#endif  // HAVE_TDM

// ---------------------------------------------------------------------------
// WMMA fragment loads (bf16 16x16x32 layouts per CDNA5 ISA §7.12.2)
// ---------------------------------------------------------------------------
static __device__ __forceinline__ v16bf load_a_frag(const __bf16* tile, int lane) {
  int m = lane & 15, h = lane >> 4;
  v8bf x = *(const v8bf*)(tile + m * LDSA_STRIDE + h * 8);
  v8bf y = *(const v8bf*)(tile + m * LDSA_STRIDE + 16 + h * 8);
  v16bf r;
#pragma unroll
  for (int j = 0; j < 8; ++j) { r[j] = x[j]; r[8 + j] = y[j]; }
  return r;
}
static __device__ __forceinline__ v16bf load_b_frag(const __bf16* tileT, int lane) {
  int n = lane & 15, h = lane >> 4;
  v8bf x = *(const v8bf*)(tileT + n * LDSB_STRIDE + h * 8);
  v8bf y = *(const v8bf*)(tileT + n * LDSB_STRIDE + 16 + h * 8);
  v16bf r;
#pragma unroll
  for (int j = 0; j < 8; ++j) { r[j] = x[j]; r[8 + j] = y[j]; }
  return r;
}

#define WMMA_BF16(a, b, c) \
  __builtin_amdgcn_wmma_f32_16x16x32_bf16(false, (a), false, (b), (short)0, (c), false, false)

// ---------------------------------------------------------------------------
// Manual staging helpers (128-bit wide)
// ---------------------------------------------------------------------------
static __device__ __forceinline__ void stage_a_manual(
    const __bf16* __restrict__ A, int lda, int rb, int k0,
    __bf16* __restrict__ ldsA, int tid) {
#pragma unroll
  for (int i = tid; i < BM * BK / 8; i += BDIM) {
    int r = i >> 2, c8 = i & 3;
    uint4v v = *(const uint4v*)(A + (size_t)(rb + r) * lda + k0 + 8 * c8);
    *(uint4v*)(&ldsA[r * LDSA_STRIDE + 8 * c8]) = v;
  }
}
// B tile [BK x BN] -> K-transposed LDS [n][k]; exactly 1 iter for 256 threads.
static __device__ __forceinline__ void stage_b_trans(
    const __bf16* __restrict__ W, int ldw, int k0, int nb,
    __bf16* __restrict__ ldsB, int tid) {
  int k = tid >> 3, n8 = tid & 7;
  uint4v v = *(const uint4v*)(W + (size_t)(k0 + k) * ldw + nb + 8 * n8);
#pragma unroll
  for (int w = 0; w < 4; ++w) {
    unsigned q = (unsigned)v[w];
    ldsB[(8 * n8 + 2 * w) * LDSB_STRIDE + k] =
        __builtin_bit_cast(__bf16, (unsigned short)(q & 0xffffu));
    ldsB[(8 * n8 + 2 * w + 1) * LDSB_STRIDE + k] =
        __builtin_bit_cast(__bf16, (unsigned short)(q >> 16));
  }
}

// ---------------------------------------------------------------------------
// Weight casting / repacking
// ---------------------------------------------------------------------------
__global__ void cast_f32_bf16_kernel(const float* __restrict__ in,
                                     __bf16* __restrict__ out, int n) {
  int i = blockIdx.x * blockDim.x + threadIdx.x;
  if (i < n) out[i] = f2bf(in[i]);
}

// Wc [N=256][M=256][3] -> Wcb [t][n][m] bf16
__global__ void conv_w_repack_kernel(const float* __restrict__ Wc,
                                     __bf16* __restrict__ Wcb) {
  int i = blockIdx.x * blockDim.x + threadIdx.x;
  if (i < 3 * 256 * 256) {
    int t = i >> 16, rem = i & 65535;
    int n = rem >> 8, m = rem & 255;
    Wcb[i] = f2bf(Wc[(n * 256 + m) * 3 + t]);
  }
}

// ---------------------------------------------------------------------------
// LayerNorm over C=512, one wave per row, output bf16
// ---------------------------------------------------------------------------
__global__ __launch_bounds__(BDIM) void ln_kernel(const float* __restrict__ x,
                                                  const float* __restrict__ g,
                                                  const float* __restrict__ b,
                                                  __bf16* __restrict__ xn) {
  int wave = threadIdx.x >> 5, lane = threadIdx.x & 31;
  int row = blockIdx.x * 8 + wave;
  const float* xr = x + (size_t)row * 512;
  float vals[16], s = 0.f, ss = 0.f;
#pragma unroll
  for (int j = 0; j < 16; ++j) {
    float v = xr[lane + 32 * j];
    vals[j] = v; s += v; ss += v * v;
  }
#pragma unroll
  for (int off = 16; off > 0; off >>= 1) {
    s += __shfl_xor(s, off, 32);
    ss += __shfl_xor(ss, off, 32);
  }
  float mu = s * (1.f / 512.f);
  float var = ss * (1.f / 512.f) - mu * mu;
  float rs = rsqrtf(var + 1e-5f);
#pragma unroll
  for (int j = 0; j < 16; ++j) {
    int c = lane + 32 * j;
    xn[(size_t)row * 512 + c] = f2bf((vals[j] - mu) * rs * g[c] + b[c]);
  }
}

// ---------------------------------------------------------------------------
// GEMM: h = relu(xn @ Ws + bs), channel-shift fused into store -> hs (bf16)
// ---------------------------------------------------------------------------
__global__ __launch_bounds__(BDIM) void gemm_s_shift_kernel(
    const __bf16* __restrict__ A, const __bf16* __restrict__ W,
    const float* __restrict__ bs, __bf16* __restrict__ hs) {
  __shared__ __bf16 ldsA[2][BM * LDSA_STRIDE];
  __shared__ __bf16 ldsB[2][BN * LDSB_STRIDE];
  const int tid = threadIdx.x, lane = tid & 31, wave = tid >> 5;
  const int wm = wave >> 1, wn = wave & 1;
  const int rb = blockIdx.y * BM, nb = blockIdx.x * BN;
  const int NK = 512 / BK;

  // stage chunk 0 into buffer 0
#if defined(HAVE_TDM)
  if (tid < 32)
    tdm_load_bf16(lds_off_of(&ldsA[0][0]), A + (size_t)rb * 512,
                  512, 16384, 0, 512, 0, BK, BM, 0);
#else
  stage_a_manual(A, 512, rb, 0, &ldsA[0][0], tid);
#endif
  stage_b_trans(W, 512, 0, nb, &ldsB[0][0], tid);

  v8f acc[2][2] = {};
  for (int k = 0; k < NK; ++k) {
    const int cur = k & 1, nxt = cur ^ 1;
    if (k + 1 < NK) {
      const int k0n = (k + 1) * BK;
#if defined(HAVE_TDM)
      if (tid < 32)
        tdm_load_bf16(lds_off_of(&ldsA[nxt][0]), A + (size_t)rb * 512 + k0n,
                      512, 16384, 0, 512, 0, BK, BM, 0);
#else
      stage_a_manual(A, 512, rb, k0n, &ldsA[nxt][0], tid);
#endif
      stage_b_trans(W, 512, k0n, nb, &ldsB[nxt][0], tid);
      if (lane == 0 && k + 2 < NK)
        __builtin_prefetch(W + (size_t)(k + 2) * BK * 512 + nb, 0, 3);
    }
#if defined(HAVE_TDM)
    if (tid < 32) {
      if (k + 1 < NK) __builtin_amdgcn_s_wait_tensorcnt(1);
      else            __builtin_amdgcn_s_wait_tensorcnt(0);
    }
#endif
    __syncthreads();
    v16bf af[2], bf[2];
#pragma unroll
    for (int mt = 0; mt < 2; ++mt)
      af[mt] = load_a_frag(&ldsA[cur][(wm * 32 + 16 * mt) * LDSA_STRIDE], lane);
#pragma unroll
    for (int nt = 0; nt < 2; ++nt)
      bf[nt] = load_b_frag(&ldsB[cur][(wn * 32 + 16 * nt) * LDSB_STRIDE], lane);
#pragma unroll
    for (int mt = 0; mt < 2; ++mt)
#pragma unroll
      for (int nt = 0; nt < 2; ++nt)
        acc[mt][nt] = WMMA_BF16(af[mt], bf[nt], acc[mt][nt]);
    __syncthreads();
  }
  const int h = lane >> 4, cl = lane & 15;
#pragma unroll
  for (int mt = 0; mt < 2; ++mt)
#pragma unroll
    for (int nt = 0; nt < 2; ++nt) {
      int gcol = nb + wn * 32 + 16 * nt + cl;
#pragma unroll
      for (int r = 0; r < 8; ++r) {
        int grow = rb + wm * 32 + 16 * mt + r + 8 * h;
        float v = acc[mt][nt][r] + bs[gcol];
        v = v > 0.f ? v : 0.f;
        int nn = grow & 255;
        int sh = (nn < 86) ? -1 : ((nn < 172) ? 0 : 1);
        int dc = gcol + sh;
        if (dc >= 0 && dc < 512) hs[(size_t)grow * 512 + dc] = f2bf(v);
        if (sh == -1 && gcol == 511) hs[(size_t)grow * 512 + 511] = f2bf(0.f);
        if (sh == 1 && gcol == 0)    hs[(size_t)grow * 512 + 0]   = f2bf(0.f);
      }
    }
}

// ---------------------------------------------------------------------------
// Conv1d over N as 3 GEMMs; one 3D TDM fetches all 3 tap A-tiles per chunk.
// ---------------------------------------------------------------------------
__global__ __launch_bounds__(BDIM) void conv_n_kernel(
    const __bf16* __restrict__ hs, const __bf16* __restrict__ Wcb,
    const float* __restrict__ bc, __bf16* __restrict__ convo) {
  __shared__ __bf16 ldsA[2][3][BM * LDSA_STRIDE];
  __shared__ __bf16 ldsB[2][66 * LDSB_STRIDE];       // cols cb-1 .. cb+64
  const int tid = threadIdx.x, lane = tid & 31, wave = tid >> 5;
  const int wm = wave >> 1, wn = wave & 1;
  const int bIdx = blockIdx.z;
  const int rb = blockIdx.y * BM;                    // n rows
  const int cb = blockIdx.x * BN;                    // c cols
  const int NK = 256 / BK;

  auto stage_a = [&](int k0, int buf) {
#if defined(HAVE_TDM)
    if (tid < 32)
      tdm_load_bf16(lds_off_of(&ldsA[buf][0][0]),
                    Wcb + (size_t)rb * 256 + k0,
                    256, 256, 3, 256, 65536, BK, BM, 3);
#else
#pragma unroll
    for (int i = tid; i < 3 * BM * BK / 8; i += BDIM) {
      int t = i >> 9, j = i & 511;
      int r = j >> 2, c8 = j & 3;
      uint4v v = *(const uint4v*)(Wcb + (size_t)t * 65536 +
                                  (size_t)(rb + r) * 256 + k0 + 8 * c8);
      *(uint4v*)(&ldsA[buf][t][r * LDSA_STRIDE + 8 * c8]) = v;
    }
#endif
  };
  auto stage_b = [&](int k0, int buf) {
    for (int i = tid; i < 66 * BK; i += BDIM) {
      int kk = i / 66, c = i % 66;
      int gc = cb - 1 + c;
      __bf16 v = f2bf(0.f);
      if (gc >= 0 && gc < 512)
        v = hs[((size_t)bIdx * 256 + k0 + kk) * 512 + gc];
      ldsB[buf][c * LDSB_STRIDE + kk] = v;
    }
  };

  stage_a(0, 0);
  stage_b(0, 0);
  v8f acc[2][2] = {};
  for (int k = 0; k < NK; ++k) {
    const int cur = k & 1, nxt = cur ^ 1;
    if (k + 1 < NK) { stage_a((k + 1) * BK, nxt); stage_b((k + 1) * BK, nxt); }
#if defined(HAVE_TDM)
    if (tid < 32) {
      if (k + 1 < NK) __builtin_amdgcn_s_wait_tensorcnt(1);
      else            __builtin_amdgcn_s_wait_tensorcnt(0);
    }
#endif
    __syncthreads();
#pragma unroll
    for (int t = 0; t < 3; ++t) {
      v16bf af0 = load_a_frag(&ldsA[cur][t][(wm * 32 + 0) * LDSA_STRIDE], lane);
      v16bf af1 = load_a_frag(&ldsA[cur][t][(wm * 32 + 16) * LDSA_STRIDE], lane);
#pragma unroll
      for (int nt = 0; nt < 2; ++nt) {
        v16bf bfr = load_b_frag(&ldsB[cur][(wn * 32 + 16 * nt + t) * LDSB_STRIDE], lane);
        acc[0][nt] = WMMA_BF16(af0, bfr, acc[0][nt]);
        acc[1][nt] = WMMA_BF16(af1, bfr, acc[1][nt]);
      }
    }
    __syncthreads();
  }
  const int h = lane >> 4, cl = lane & 15;
#pragma unroll
  for (int mt = 0; mt < 2; ++mt)
#pragma unroll
    for (int nt = 0; nt < 2; ++nt) {
      int gcol = cb + wn * 32 + 16 * nt + cl;
#pragma unroll
      for (int r = 0; r < 8; ++r) {
        int grow = rb + wm * 32 + 16 * mt + r + 8 * h;
        float v = acc[mt][nt][r] + bc[grow];
        convo[((size_t)bIdx * 256 + grow) * 512 + gcol] = f2bf(v);
      }
    }
}

// ---------------------------------------------------------------------------
// SE squeeze / excite
// ---------------------------------------------------------------------------
__global__ __launch_bounds__(BDIM) void rowmean_kernel(const __bf16* __restrict__ convo,
                                                       float* __restrict__ se) {
  int wave = threadIdx.x >> 5, lane = threadIdx.x & 31;
  int row = blockIdx.x * 8 + wave;
  const __bf16* xr = convo + (size_t)row * 512;
  float s = 0.f;
#pragma unroll
  for (int j = 0; j < 16; ++j) s += bf2f(xr[lane + 32 * j]);
#pragma unroll
  for (int off = 16; off > 0; off >>= 1) s += __shfl_xor(s, off, 32);
  if (lane == 0) se[row] = s * (1.f / 512.f);
}

__global__ __launch_bounds__(BDIM) void se_kernel(
    const float* __restrict__ se, const float* __restrict__ We1,
    const float* __restrict__ be1, const float* __restrict__ We2,
    const float* __restrict__ be2, float* __restrict__ ex) {
  __shared__ float sSe[256];
  __shared__ float sH[32];
  int b = blockIdx.x, tid = threadIdx.x;
  sSe[tid] = se[b * 256 + tid];
  __syncthreads();
  if (tid < 32) {
    float a = be1[tid];
    for (int m = 0; m < 256; ++m) a += sSe[m] * We1[m * 32 + tid];
    sH[tid] = a > 0.f ? a : 0.f;
  }
  __syncthreads();
  float a = be2[tid];
#pragma unroll
  for (int j = 0; j < 32; ++j) a += sH[j] * We2[j * 256 + tid];
  ex[b * 256 + tid] = 1.f / (1.f + expf(-a));
}

// ---------------------------------------------------------------------------
// FFN1: g = relu((conv * ex) @ Wf1 + bf1); gate applied while staging A
// ---------------------------------------------------------------------------
__global__ __launch_bounds__(BDIM) void ffn1_kernel(
    const __bf16* __restrict__ convo, const float* __restrict__ ex,
    const __bf16* __restrict__ W, const float* __restrict__ bf1,
    __bf16* __restrict__ g) {
  __shared__ __bf16 ldsA[BM * LDSA_STRIDE];
  __shared__ __bf16 ldsB[BN * LDSB_STRIDE];
  const int tid = threadIdx.x, lane = tid & 31, wave = tid >> 5;
  const int wm = wave >> 1, wn = wave & 1;
  const int rb = blockIdx.y * BM, nb = blockIdx.x * BN;

  v8f acc[2][2] = {};
  for (int k0 = 0; k0 < 512; k0 += BK) {
#pragma unroll
    for (int i = tid; i < BM * BK / 8; i += BDIM) {
      int r = i >> 2, c8 = i & 3;
      int row = rb + r;
      uint4v v = *(const uint4v*)(convo + (size_t)row * 512 + k0 + 8 * c8);
      float e = ex[row];
      uint4v o;
#pragma unroll
      for (int w = 0; w < 4; ++w) {
        unsigned q = (unsigned)v[w];
        float fl = __uint_as_float((q & 0xffffu) << 16) * e;
        float fh = __uint_as_float((q >> 16) << 16) * e;
        o[w] = (unsigned)__builtin_bit_cast(unsigned short, f2bf(fl)) |
               ((unsigned)__builtin_bit_cast(unsigned short, f2bf(fh)) << 16);
      }
      *(uint4v*)(&ldsA[r * LDSA_STRIDE + 8 * c8]) = o;
    }
    stage_b_trans(W, 1024, k0, nb, ldsB, tid);
    if (lane == 0 && k0 + BK < 512)
      __builtin_prefetch(W + (size_t)(k0 + BK) * 1024 + nb, 0, 3);
    __syncthreads();
    v16bf af[2], bf[2];
#pragma unroll
    for (int mt = 0; mt < 2; ++mt)
      af[mt] = load_a_frag(&ldsA[(wm * 32 + 16 * mt) * LDSA_STRIDE], lane);
#pragma unroll
    for (int nt = 0; nt < 2; ++nt)
      bf[nt] = load_b_frag(&ldsB[(wn * 32 + 16 * nt) * LDSB_STRIDE], lane);
#pragma unroll
    for (int mt = 0; mt < 2; ++mt)
#pragma unroll
      for (int nt = 0; nt < 2; ++nt)
        acc[mt][nt] = WMMA_BF16(af[mt], bf[nt], acc[mt][nt]);
    __syncthreads();
  }
  const int h = lane >> 4, cl = lane & 15;
#pragma unroll
  for (int mt = 0; mt < 2; ++mt)
#pragma unroll
    for (int nt = 0; nt < 2; ++nt) {
      int gcol = nb + wn * 32 + 16 * nt + cl;
#pragma unroll
      for (int r = 0; r < 8; ++r) {
        int grow = rb + wm * 32 + 16 * mt + r + 8 * h;
        float v = acc[mt][nt][r] + bf1[gcol];
        v = v > 0.f ? v : 0.f;
        g[(size_t)grow * 1024 + gcol] = f2bf(v);
      }
    }
}

// ---------------------------------------------------------------------------
// FFN2: out = g @ Wf2 + bf2 + x (fp32 residual); TDM-staged A
// ---------------------------------------------------------------------------
__global__ __launch_bounds__(BDIM) void ffn2_kernel(
    const __bf16* __restrict__ g, const __bf16* __restrict__ W,
    const float* __restrict__ bf2, const float* __restrict__ xres,
    float* __restrict__ out) {
  __shared__ __bf16 ldsA[2][BM * LDSA_STRIDE];
  __shared__ __bf16 ldsB[2][BN * LDSB_STRIDE];
  const int tid = threadIdx.x, lane = tid & 31, wave = tid >> 5;
  const int wm = wave >> 1, wn = wave & 1;
  const int rb = blockIdx.y * BM, nb = blockIdx.x * BN;
  const int NK = 1024 / BK;

#if defined(HAVE_TDM)
  if (tid < 32)
    tdm_load_bf16(lds_off_of(&ldsA[0][0]), g + (size_t)rb * 1024,
                  1024, 16384, 0, 1024, 0, BK, BM, 0);
#else
  stage_a_manual(g, 1024, rb, 0, &ldsA[0][0], tid);
#endif
  stage_b_trans(W, 512, 0, nb, &ldsB[0][0], tid);

  v8f acc[2][2] = {};
  for (int k = 0; k < NK; ++k) {
    const int cur = k & 1, nxt = cur ^ 1;
    if (k + 1 < NK) {
      const int k0n = (k + 1) * BK;
#if defined(HAVE_TDM)
      if (tid < 32)
        tdm_load_bf16(lds_off_of(&ldsA[nxt][0]), g + (size_t)rb * 1024 + k0n,
                      1024, 16384, 0, 1024, 0, BK, BM, 0);
#else
      stage_a_manual(g, 1024, rb, k0n, &ldsA[nxt][0], tid);
#endif
      stage_b_trans(W, 512, k0n, nb, &ldsB[nxt][0], tid);
      if (lane == 0 && k + 2 < NK)
        __builtin_prefetch(W + (size_t)(k + 2) * BK * 512 + nb, 0, 3);
    }
#if defined(HAVE_TDM)
    if (tid < 32) {
      if (k + 1 < NK) __builtin_amdgcn_s_wait_tensorcnt(1);
      else            __builtin_amdgcn_s_wait_tensorcnt(0);
    }
#endif
    __syncthreads();
    v16bf af[2], bf[2];
#pragma unroll
    for (int mt = 0; mt < 2; ++mt)
      af[mt] = load_a_frag(&ldsA[cur][(wm * 32 + 16 * mt) * LDSA_STRIDE], lane);
#pragma unroll
    for (int nt = 0; nt < 2; ++nt)
      bf[nt] = load_b_frag(&ldsB[cur][(wn * 32 + 16 * nt) * LDSB_STRIDE], lane);
#pragma unroll
    for (int mt = 0; mt < 2; ++mt)
#pragma unroll
      for (int nt = 0; nt < 2; ++nt)
        acc[mt][nt] = WMMA_BF16(af[mt], bf[nt], acc[mt][nt]);
    __syncthreads();
  }
  const int h = lane >> 4, cl = lane & 15;
#pragma unroll
  for (int mt = 0; mt < 2; ++mt)
#pragma unroll
    for (int nt = 0; nt < 2; ++nt) {
      int gcol = nb + wn * 32 + 16 * nt + cl;
#pragma unroll
      for (int r = 0; r < 8; ++r) {
        int grow = rb + wm * 32 + 16 * mt + r + 8 * h;
        out[(size_t)grow * 512 + gcol] =
            acc[mt][nt][r] + bf2[gcol] + xres[(size_t)grow * 512 + gcol];
      }
    }
}

// ---------------------------------------------------------------------------
extern "C" void kernel_launch(void* const* d_in, const int* in_sizes, int n_in,
                              void* d_out, int out_size, void* d_ws, size_t ws_size,
                              hipStream_t stream) {
  (void)in_sizes; (void)n_in; (void)out_size; (void)ws_size;
  const float* x    = (const float*)d_in[0];
  const float* ln_g = (const float*)d_in[1];
  const float* ln_b = (const float*)d_in[2];
  const float* Ws   = (const float*)d_in[3];
  const float* bs   = (const float*)d_in[4];
  const float* Wc   = (const float*)d_in[5];
  const float* bc   = (const float*)d_in[6];
  const float* We1  = (const float*)d_in[7];
  const float* be1  = (const float*)d_in[8];
  const float* We2  = (const float*)d_in[9];
  const float* be2  = (const float*)d_in[10];
  const float* Wf1  = (const float*)d_in[11];
  const float* bf1  = (const float*)d_in[12];
  const float* Wf2  = (const float*)d_in[13];
  const float* bf2  = (const float*)d_in[14];
  float* out = (float*)d_out;

  char* ws = (char*)d_ws;
  size_t off = 0;
  auto alloc = [&](size_t bytes) {
    char* p = ws + off;
    off += (bytes + 255) & ~(size_t)255;
    return p;
  };
  const size_t ROWS = 16384;  // B*N
  __bf16* xn    = (__bf16*)alloc(ROWS * 512 * 2);
  __bf16* hs    = (__bf16*)alloc(ROWS * 512 * 2);
  __bf16* convo = (__bf16*)alloc(ROWS * 512 * 2);
  __bf16* gbuf  = (__bf16*)alloc(ROWS * 1024 * 2);
  __bf16* Wsb   = (__bf16*)alloc(512 * 512 * 2);
  __bf16* Wcb   = (__bf16*)alloc(3 * 256 * 256 * 2);
  __bf16* Wf1b  = (__bf16*)alloc(512 * 1024 * 2);
  __bf16* Wf2b  = (__bf16*)alloc(1024 * 512 * 2);
  float*  se    = (float*)alloc(ROWS * 4);
  float*  exg   = (float*)alloc(ROWS * 4);

  cast_f32_bf16_kernel<<<(512 * 512 + 255) / 256, 256, 0, stream>>>(Ws, Wsb, 512 * 512);
  cast_f32_bf16_kernel<<<(512 * 1024 + 255) / 256, 256, 0, stream>>>(Wf1, Wf1b, 512 * 1024);
  cast_f32_bf16_kernel<<<(1024 * 512 + 255) / 256, 256, 0, stream>>>(Wf2, Wf2b, 1024 * 512);
  conv_w_repack_kernel<<<(3 * 256 * 256 + 255) / 256, 256, 0, stream>>>(Wc, Wcb);

  ln_kernel<<<ROWS / 8, BDIM, 0, stream>>>(x, ln_g, ln_b, xn);
  gemm_s_shift_kernel<<<dim3(512 / BN, ROWS / BM), BDIM, 0, stream>>>(xn, Wsb, bs, hs);
  conv_n_kernel<<<dim3(512 / BN, 256 / BM, 64), BDIM, 0, stream>>>(hs, Wcb, bc, convo);
  rowmean_kernel<<<ROWS / 8, BDIM, 0, stream>>>(convo, se);
  se_kernel<<<64, BDIM, 0, stream>>>(se, We1, be1, We2, be2, exg);
  ffn1_kernel<<<dim3(1024 / BN, ROWS / BM), BDIM, 0, stream>>>(convo, exg, Wf1b, bf1, gbuf);
  ffn2_kernel<<<dim3(512 / BN, ROWS / BM), BDIM, 0, stream>>>(gbuf, Wf2b, bf2, x, out);
}
